// ResidualPathME_2946347565628
// MI455X (gfx1250) — matrix operand order repaired and checked by reading, not compiled, not run on hardware
//
#include <hip/hip_runtime.h>

// ---------------------------------------------------------------------------
// Types for CDNA5 WMMA (wave32): V_WMMA_F32_16X16X32_BF16
// ---------------------------------------------------------------------------
typedef __attribute__((ext_vector_type(16))) __bf16 bf16x16;
typedef __attribute__((ext_vector_type(8)))  __bf16 bf16x8;
typedef __attribute__((ext_vector_type(8)))  float  f32x8;

static __device__ __forceinline__ bf16x16 ld_bf16x16_pair(const __bf16* p0, const __bf16* p1) {
  bf16x8 lo = *(const bf16x8*)p0;   // 16-byte load
  bf16x8 hi = *(const bf16x8*)p1;   // 16-byte load
  return __builtin_shufflevector(lo, hi, 0,1,2,3,4,5,6,7,8,9,10,11,12,13,14,15);
}

// ---------------------------------------------------------------------------
// Small prep kernels
// ---------------------------------------------------------------------------
__global__ void zero_kernel(float* p, int n) {
  int i = blockIdx.x * blockDim.x + threadIdx.x;
  if (i < n) p[i] = 0.0f;
}

__global__ void cvt_bf16_kernel(const float* __restrict__ in, __bf16* __restrict__ out, int n) {
  int i = blockIdx.x * blockDim.x + threadIdx.x;
  if (i < n) out[i] = (__bf16)in[i];
}

// W [T][Ci][Co] f32  ->  WT [T][Co][Ci] bf16  (B-matrix column-contiguous in K)
__global__ void transpose_w_kernel(const float* __restrict__ W, __bf16* __restrict__ WT,
                                   int T, int Ci, int Co) {
  int i = blockIdx.x * blockDim.x + threadIdx.x;
  int total = T * Ci * Co;
  if (i >= total) return;
  int co = i % Co;
  int ci = (i / Co) % Ci;
  int t  = i / (Co * Ci);
  WT[((size_t)t * Co + co) * Ci + ci] = (__bf16)W[i];
}

// scale/shift = fold(gamma, beta, mean, var)
__global__ void finalize_bn_kernel(const float* __restrict__ sum, const float* __restrict__ sq,
                                   const float* __restrict__ g, const float* __restrict__ b,
                                   float* __restrict__ scale, float* __restrict__ shift,
                                   int C, float inv_count) {
  int c = blockIdx.x * blockDim.x + threadIdx.x;
  if (c >= C) return;
  float mu  = sum[c] * inv_count;
  float var = sq[c] * inv_count - mu * mu;
  float sc  = g[c] * rsqrtf(var + 1e-5f);
  scale[c] = sc;
  shift[c] = b[c] - mu * sc;
}

// out = relu(in*scale[c] + shift[c]); supports in-place (in == out)
__global__ void bn_apply_kernel(const float* __restrict__ in, float* __restrict__ out,
                                const float* __restrict__ scale, const float* __restrict__ shift,
                                int C, long total) {
  long i = (long)blockIdx.x * blockDim.x + threadIdx.x;
  if (i >= total) return;
  int c = (int)(i % C);
  float v = in[i] * scale[c] + shift[c];
  out[i] = v > 0.0f ? v : 0.0f;
}

// ---------------------------------------------------------------------------
// conv1: y1_raw[N,64] = sum_k gather(feats_bf, nbr1[k]) @ W1T[k]   (bf16 WMMA)
// One wave = 16-point x 64-out tile, K=32 per WMMA.
//  - per-tap weight tiles double-buffered through LDS (block-cooperative)
//  - A gather software-pipelined one tap ahead (branchless, clamped index)
//  - all 4 B fragments loaded into distinct regs before the WMMA group so the
//    ds_loads overlap instead of ping-ponging one register group
// ---------------------------------------------------------------------------
__global__ __launch_bounds__(256)
void conv1_wmma_kernel(const __bf16* __restrict__ fb, const int* __restrict__ nbr,
                       const __bf16* __restrict__ WT, float* __restrict__ yraw,
                       float* __restrict__ gsum, float* __restrict__ gsq, int N) {
  __shared__ __align__(16) __bf16 swt[2][64 * 32];   // 2 x 4KB weight tiles
  __shared__ float ssum[64], ssq[64];
  const int t = threadIdx.x;
  if (t < 64) { ssum[t] = 0.0f; ssq[t] = 0.0f; }
  if (t < 128) ((uint4*)&swt[0][0])[t] = ((const uint4*)WT)[t];  // preload tap 0
  __syncthreads();

  const int lane = t & 31;
  const int wave = t >> 5;
  const int base = blockIdx.x * 128 + wave * 16;
  const int m    = lane & 15;        // tile row (point) this lane serves
  const int hi   = lane >> 4;        // lane half
  const int klo  = hi * 8;           // unused (A fragment also indexes with kb)
  const int kb   = hi * 16;          // B-fragment K base
  const int gmrow  = base + m;
  const int gmclmp = gmrow < N ? gmrow : N - 1;   // clamped: loads always legal
  const bool rowok = gmrow < N;
  const bf16x16 zfrag = {};

  // branchless gather of one A fragment for tap k
  auto gatherA = [&](int k) -> bf16x16 {
    int idx = nbr[(size_t)k * N + gmclmp];
    idx = rowok ? idx : -1;
    int idxr = idx < 0 ? 0 : idx;
    const __bf16* row = fb + (size_t)idxr * 32;
    bf16x16 a = ld_bf16x16_pair(row + kb, row + kb + 8);
    return (idx >= 0) ? a : zfrag;
  };

  f32x8 acc[4] = {};
  bf16x16 a_cur = gatherA(0);
  for (int k = 0; k < 27; ++k) {
    const int cur = k & 1, nxt = cur ^ 1;
    const bool more = (k + 1 < 27);
    // --- pipeline: start next tap's A gather + weight fetch ---
    bf16x16 a_next = zfrag;
    if (more) a_next = gatherA(k + 1);
    uint4 wreg;
    if (more && t < 128) wreg = ((const uint4*)(WT + (size_t)(k + 1) * 2048))[t];
    // --- load all B fragments of this tap, then WMMA group ---
    const __bf16* wb = &swt[cur][0];
    bf16x16 b0 = ld_bf16x16_pair(wb + (0 * 16 + m) * 32 + kb, wb + (0 * 16 + m) * 32 + kb + 8);
    bf16x16 b1 = ld_bf16x16_pair(wb + (1 * 16 + m) * 32 + kb, wb + (1 * 16 + m) * 32 + kb + 8);
    bf16x16 b2 = ld_bf16x16_pair(wb + (2 * 16 + m) * 32 + kb, wb + (2 * 16 + m) * 32 + kb + 8);
    bf16x16 b3 = ld_bf16x16_pair(wb + (3 * 16 + m) * 32 + kb, wb + (3 * 16 + m) * 32 + kb + 8);
    acc[0] = __builtin_amdgcn_wmma_f32_16x16x32_bf16(false, a_cur, false, b0, (short)0, acc[0], false, false);
    acc[1] = __builtin_amdgcn_wmma_f32_16x16x32_bf16(false, a_cur, false, b1, (short)0, acc[1], false, false);
    acc[2] = __builtin_amdgcn_wmma_f32_16x16x32_bf16(false, a_cur, false, b2, (short)0, acc[2], false, false);
    acc[3] = __builtin_amdgcn_wmma_f32_16x16x32_bf16(false, a_cur, false, b3, (short)0, acc[3], false, false);
    // --- commit next tap's weights, flip buffers ---
    if (more && t < 128) ((uint4*)&swt[nxt][0])[t] = wreg;
    __syncthreads();
    a_cur = a_next;
  }

#pragma unroll
  for (int n0 = 0; n0 < 4; ++n0) {
    const int n = n0 * 16 + m;
    float s = 0.0f, q = 0.0f;
#pragma unroll
    for (int r = 0; r < 8; ++r) {
      const int gm = base + r + hi * 8;   // C layout: lane half selects M 0-7 / 8-15
      if (gm < N) {
        float v = acc[n0][r];
        yraw[(size_t)gm * 64 + n] = v;
        s += v; q += v * v;
      }
    }
    atomicAdd(&ssum[n], s);
    atomicAdd(&ssq[n], q);
  }
  __syncthreads();
  if (t < 64) { atomicAdd(&gsum[t], ssum[t]); atomicAdd(&gsq[t], ssq[t]); }
}

// ---------------------------------------------------------------------------
// conv2: Cin=64 -> two K=32 WMMA steps per tap; 8KB weight tiles
// double-buffered through LDS.  Same pipelining as conv1.
// Writes raw output (pre-BN) to d_out.
// ---------------------------------------------------------------------------
__global__ __launch_bounds__(256)
void conv2_wmma_kernel(const __bf16* __restrict__ sb, const int* __restrict__ nbr,
                       const __bf16* __restrict__ WT, float* __restrict__ yraw,
                       float* __restrict__ gsum, float* __restrict__ gsq, int N) {
  __shared__ __align__(16) __bf16 swt[2][64 * 64];   // 2 x 8KB weight tiles
  __shared__ float ssum[64], ssq[64];
  const int t = threadIdx.x;
  if (t < 64) { ssum[t] = 0.0f; ssq[t] = 0.0f; }
  ((uint4*)&swt[0][0])[t] = ((const uint4*)WT)[t];   // preload tap 0 (256 uint4)
  __syncthreads();

  const int lane = t & 31;
  const int wave = t >> 5;
  const int base = blockIdx.x * 128 + wave * 16;
  const int m    = lane & 15;
  const int hi   = lane >> 4;
  const int klo  = hi * 8;
  const int kb   = hi * 16;
  const int gmrow  = base + m;
  const int gmclmp = gmrow < N ? gmrow : N - 1;
  const bool rowok = gmrow < N;
  const bf16x16 zfrag = {};

  struct APair { bf16x16 a0, a1; };
  auto gatherA = [&](int k) -> APair {
    int idx = nbr[(size_t)k * N + gmclmp];
    idx = rowok ? idx : -1;
    int idxr = idx < 0 ? 0 : idx;
    const __bf16* row = sb + (size_t)idxr * 64;
    APair p;
    p.a0 = ld_bf16x16_pair(row + kb,      row + kb + 8);   // K 0..31
    p.a1 = ld_bf16x16_pair(row + 32 + kb, row + 32 + kb + 8);   // K 32..63
    p.a0 = (idx >= 0) ? p.a0 : zfrag;
    p.a1 = (idx >= 0) ? p.a1 : zfrag;
    return p;
  };

  f32x8 acc[4] = {};
  APair a_cur = gatherA(0);
  for (int k = 0; k < 27; ++k) {
    const int cur = k & 1, nxt = cur ^ 1;
    const bool more = (k + 1 < 27);
    // --- pipeline: next tap's A gather + weight fetch ---
    APair a_next;
    a_next.a0 = zfrag; a_next.a1 = zfrag;
    if (more) a_next = gatherA(k + 1);
    uint4 wreg;
    if (more) wreg = ((const uint4*)(WT + (size_t)(k + 1) * 4096))[t];
    // --- per output tile: load both B fragments, then both WMMAs ---
#pragma unroll
    for (int n0 = 0; n0 < 4; ++n0) {
      const __bf16* col = &swt[cur][(n0 * 16 + m) * 64];
      bf16x16 b0 = ld_bf16x16_pair(col + kb,      col + kb + 8);
      bf16x16 b1 = ld_bf16x16_pair(col + 32 + kb, col + 32 + kb + 8);
      acc[n0] = __builtin_amdgcn_wmma_f32_16x16x32_bf16(
          false, a_cur.a0, false, b0, (short)0, acc[n0], false, false);
      acc[n0] = __builtin_amdgcn_wmma_f32_16x16x32_bf16(
          false, a_cur.a1, false, b1, (short)0, acc[n0], false, false);
    }
    // --- commit next tap's weights, flip buffers ---
    if (more) ((uint4*)&swt[nxt][0])[t] = wreg;
    __syncthreads();
    a_cur = a_next;
  }

#pragma unroll
  for (int n0 = 0; n0 < 4; ++n0) {
    const int n = n0 * 16 + m;
    float s = 0.0f, q = 0.0f;
#pragma unroll
    for (int r = 0; r < 8; ++r) {
      const int gm = base + r + hi * 8;
      if (gm < N) {
        float v = acc[n0][r];
        yraw[(size_t)gm * 64 + n] = v;
        s += v; q += v * v;
      }
    }
    atomicAdd(&ssum[n], s);
    atomicAdd(&ssq[n], q);
  }
  __syncthreads();
  if (t < 64) { atomicAdd(&gsum[t], ssum[t]); atomicAdd(&gsq[t], ssq[t]); }
}

// ---------------------------------------------------------------------------
// apply BN1 + relu -> x ; SE per-batch pooling partials ; att_raw = x @ Wp1
// (+ its BN stat partials). One thread per point.
// ---------------------------------------------------------------------------
__global__ __launch_bounds__(256)
void apply_bn1_kernel(const float* __restrict__ yraw,
                      const float* __restrict__ scale, const float* __restrict__ shift,
                      const int* __restrict__ bidx, const float* __restrict__ Wp1,
                      float* __restrict__ x, float* __restrict__ attraw,
                      float* __restrict__ gpool, float* __restrict__ gasum,
                      float* __restrict__ gasq, int N) {
  __shared__ float sWp[64 * 8];
  __shared__ float ssc[64], ssh[64];
  __shared__ float spool[2 * 64];
  __shared__ float sas[8], sasq[8];
  const int t = threadIdx.x;
  for (int i = t; i < 512; i += 256) sWp[i] = Wp1[i];
  if (t < 64) { ssc[t] = scale[t]; ssh[t] = shift[t]; }
  if (t < 128) spool[t] = 0.0f;
  if (t < 8) { sas[t] = 0.0f; sasq[t] = 0.0f; }
  __syncthreads();

  const int n = blockIdx.x * 256 + t;
  if (n < N) {
    const int bi = bidx[n];
    const float* yr = yraw + (size_t)n * 64;
    float* xr = x + (size_t)n * 64;
    float av[8] = {};
    for (int c = 0; c < 64; ++c) {
      float v = yr[c] * ssc[c] + ssh[c];
      v = v > 0.0f ? v : 0.0f;
      xr[c] = v;
      atomicAdd(&spool[bi * 64 + c], v);
#pragma unroll
      for (int j = 0; j < 8; ++j) av[j] += v * sWp[c * 8 + j];
    }
#pragma unroll
    for (int j = 0; j < 8; ++j) {
      attraw[(size_t)n * 8 + j] = av[j];
      atomicAdd(&sas[j], av[j]);
      atomicAdd(&sasq[j], av[j] * av[j]);
    }
  }
  __syncthreads();
  if (t < 128) atomicAdd(&gpool[t], spool[t]);
  if (t < 8) { atomicAdd(&gasum[t], sas[t]); atomicAdd(&gasq[t], sasq[t]); }
}

// ---------------------------------------------------------------------------
// SE head: y[2,64] = sigmoid(relu(pool/cnt @ Wse1 + bse1) @ Wse2 + bse2)
// ---------------------------------------------------------------------------
__global__ void se_kernel(const float* __restrict__ pool,
                          const float* __restrict__ Wse1, const float* __restrict__ bse1,
                          const float* __restrict__ Wse2, const float* __restrict__ bse2,
                          float* __restrict__ sey, float inv_cnt) {
  __shared__ float h[16];
  const int t = threadIdx.x;
  if (t < 16) {
    int b = t >> 3, j = t & 7;
    float a = bse1[j];
    for (int c = 0; c < 64; ++c) a += pool[b * 64 + c] * inv_cnt * Wse1[c * 8 + j];
    h[t] = a > 0.0f ? a : 0.0f;
  }
  __syncthreads();
  if (t < 128) {
    int b = t >> 6, c = t & 63;
    float a = bse2[c];
#pragma unroll
    for (int j = 0; j < 8; ++j) a += h[b * 8 + j] * Wse2[j * 64 + c];
    sey[t] = 1.0f / (1.0f + expf(-a));
  }
}

// ---------------------------------------------------------------------------
// 4x dilated 27-tap 8->8 sparse convs on att. Tiny channels: pure VALU with
// all weights staged in LDS. Accumulates BN stats (32 channels).
// ---------------------------------------------------------------------------
__global__ __launch_bounds__(256)
void dilated_kernel(const float* __restrict__ att,
                    const int* __restrict__ n1, const int* __restrict__ n2,
                    const int* __restrict__ n3, const int* __restrict__ n4,
                    const float* __restrict__ Wd1, const float* __restrict__ Wd2,
                    const float* __restrict__ Wd3, const float* __restrict__ Wd4,
                    float* __restrict__ draw, float* __restrict__ gsum,
                    float* __restrict__ gsq, int N) {
  __shared__ float sw[4 * 27 * 64];
  __shared__ float ssum[32], ssq[32];
  const int t = threadIdx.x;
  for (int i = t; i < 1728; i += 256) {
    sw[i] = Wd1[i]; sw[1728 + i] = Wd2[i]; sw[3456 + i] = Wd3[i]; sw[5184 + i] = Wd4[i];
  }
  if (t < 32) { ssum[t] = 0.0f; ssq[t] = 0.0f; }
  __syncthreads();

  const int n = blockIdx.x * 256 + t;
  if (n < N) {
    const int* nbrs[4] = {n1, n2, n3, n4};
    float acc[4][8] = {};
#pragma unroll
    for (int d = 0; d < 4; ++d) {
      const int* nb = nbrs[d];
      const float* wd = sw + d * 1728;
      for (int k = 0; k < 27; ++k) {
        int idx = nb[(size_t)k * N + n];
        if (idx < 0) continue;
        const float* ar = att + (size_t)idx * 8;
        const float* wk = wd + k * 64;
        float a0 = ar[0], a1 = ar[1], a2 = ar[2], a3 = ar[3];
        float a4 = ar[4], a5 = ar[5], a6 = ar[6], a7 = ar[7];
#pragma unroll
        for (int j = 0; j < 8; ++j)
          acc[d][j] += a0 * wk[j]      + a1 * wk[8 + j]  + a2 * wk[16 + j] + a3 * wk[24 + j]
                     + a4 * wk[32 + j] + a5 * wk[40 + j] + a6 * wk[48 + j] + a7 * wk[56 + j];
      }
    }
#pragma unroll
    for (int d = 0; d < 4; ++d)
#pragma unroll
      for (int j = 0; j < 8; ++j) {
        float v = acc[d][j];
        draw[(size_t)n * 32 + d * 8 + j] = v;
        atomicAdd(&ssum[d * 8 + j], v);
        atomicAdd(&ssq[d * 8 + j], v * v);
      }
  }
  __syncthreads();
  if (t < 32) { atomicAdd(&gsum[t], ssum[t]); atomicAdd(&gsq[t], ssq[t]); }
}

// ---------------------------------------------------------------------------
// a_raw = concat(att, bn_relu(d1..d4)) @ Wp2   (+ scalar BN stats)
// ---------------------------------------------------------------------------
__global__ __launch_bounds__(256)
void a_kernel(const float* __restrict__ att, const float* __restrict__ draw,
              const float* __restrict__ dscale, const float* __restrict__ dshift,
              const float* __restrict__ Wp2, float* __restrict__ araw,
              float* __restrict__ gs, float* __restrict__ gq, int N) {
  __shared__ float sw[40];
  __shared__ float sds[32], sdh[32];
  __shared__ float ssum, ssq;
  const int t = threadIdx.x;
  if (t < 40) sw[t] = Wp2[t];
  if (t < 32) { sds[t] = dscale[t]; sdh[t] = dshift[t]; }
  if (t == 0) { ssum = 0.0f; ssq = 0.0f; }
  __syncthreads();

  const int n = blockIdx.x * 256 + t;
  if (n < N) {
    float a = 0.0f;
    const float* ar = att + (size_t)n * 8;
#pragma unroll
    for (int j = 0; j < 8; ++j) a += ar[j] * sw[j];
    const float* dr = draw + (size_t)n * 32;
#pragma unroll
    for (int i = 0; i < 32; ++i) {
      float v = dr[i] * sds[i] + sdh[i];
      v = v > 0.0f ? v : 0.0f;
      a += v * sw[8 + i];
    }
    araw[n] = a;
    atomicAdd(&ssum, a);
    atomicAdd(&ssq, a * a);
  }
  __syncthreads();
  if (t == 0) { atomicAdd(gs, ssum); atomicAdd(gq, ssq); }
}

// ---------------------------------------------------------------------------
// s = x * (1 + se_y[batch] + relu(bn(a_raw)))  -> bf16 for conv2
// ---------------------------------------------------------------------------
__global__ __launch_bounds__(256)
void s_kernel(const float* __restrict__ x, const float* __restrict__ araw,
              const float* __restrict__ ascale, const float* __restrict__ ashift,
              const float* __restrict__ sey, const int* __restrict__ bidx,
              __bf16* __restrict__ sbf, int N) {
  const int n = blockIdx.x * 256 + threadIdx.x;
  if (n >= N) return;
  float a = araw[n] * ascale[0] + ashift[0];
  a = a > 0.0f ? a : 0.0f;
  const int bi = bidx[n];
  const float* xr = x + (size_t)n * 64;
  const float* se = sey + bi * 64;
  __bf16* sr = sbf + (size_t)n * 64;
  for (int c = 0; c < 64; ++c) sr[c] = (__bf16)(xr[c] * (1.0f + se[c] + a));
}

// ---------------------------------------------------------------------------
// Host-side orchestration
// ---------------------------------------------------------------------------
extern "C" void kernel_launch(void* const* d_in, const int* in_sizes, int n_in,
                              void* d_out, int out_size, void* d_ws, size_t ws_size,
                              hipStream_t stream) {
  const float* feats = (const float*)d_in[0];
  const int*   bidx  = (const int*)d_in[1];
  const int*   nbr1  = (const int*)d_in[2];
  const int*   nbr2  = (const int*)d_in[3];
  const int*   nbr3  = (const int*)d_in[4];
  const int*   nbr4  = (const int*)d_in[5];
  const float* W1   = (const float*)d_in[6];
  const float* g1   = (const float*)d_in[7];
  const float* b1   = (const float*)d_in[8];
  const float* Wse1 = (const float*)d_in[9];
  const float* bse1 = (const float*)d_in[10];
  const float* Wse2 = (const float*)d_in[11];
  const float* bse2 = (const float*)d_in[12];
  const float* Wp1  = (const float*)d_in[13];
  const float* gp1  = (const float*)d_in[14];
  const float* bp1  = (const float*)d_in[15];
  const float* gds[4] = {(const float*)d_in[17], (const float*)d_in[20],
                         (const float*)d_in[23], (const float*)d_in[26]};
  const float* bds[4] = {(const float*)d_in[18], (const float*)d_in[21],
                         (const float*)d_in[24], (const float*)d_in[27]};
  const float* Wd1 = (const float*)d_in[16];
  const float* Wd2 = (const float*)d_in[19];
  const float* Wd3 = (const float*)d_in[22];
  const float* Wd4 = (const float*)d_in[25];
  const float* Wp2 = (const float*)d_in[28];
  const float* gp2 = (const float*)d_in[29];
  const float* bp2 = (const float*)d_in[30];
  const float* W2  = (const float*)d_in[31];
  const float* g2  = (const float*)d_in[32];
  const float* b2  = (const float*)d_in[33];

  const int N = in_sizes[0] / 32;       // feats is [N,32]
  const float invN = 1.0f / (float)N;

  // ---- workspace carving ----
  float* wsf = (float*)d_ws;
  float* sum1 = wsf + 0,   *sq1 = wsf + 64;
  float* pool = wsf + 128;                       // [2,64]
  float* asum = wsf + 256, *asq = wsf + 264;     // att stats [8]
  float* dsum = wsf + 272, *dsq = wsf + 304;     // dilated stats [32]
  float* a2sum = wsf + 336, *a2sq = wsf + 337;   // a stats [1]
  float* sum2 = wsf + 344, *sq2 = wsf + 408;
  float* bn1s = wsf + 512, *bn1h = wsf + 576;
  float* sey  = wsf + 640;                       // [2,64]
  float* atts = wsf + 768, *atth = wsf + 776;
  float* ds_  = wsf + 784, *dh_  = wsf + 816;
  float* as_  = wsf + 848, *ah_  = wsf + 849;
  float* bn2s = wsf + 856, *bn2h = wsf + 920;

  char* basep = (char*)d_ws;
  size_t off = 4096;
  auto au = [](size_t v) { return (v + 255) & ~(size_t)255; };
  __bf16* fb   = (__bf16*)(basep + off); off = au(off + (size_t)N * 32 * 2);
  __bf16* W1T  = (__bf16*)(basep + off); off = au(off + (size_t)27 * 64 * 32 * 2);
  __bf16* W2T  = (__bf16*)(basep + off); off = au(off + (size_t)27 * 64 * 64 * 2);
  float* y1    = (float*)(basep + off);  off = au(off + (size_t)N * 64 * 4);
  float* x     = (float*)(basep + off);  off = au(off + (size_t)N * 64 * 4);
  float* attraw= (float*)(basep + off);  off = au(off + (size_t)N * 8 * 4);
  float* att   = (float*)(basep + off);  off = au(off + (size_t)N * 8 * 4);
  float* draw  = (float*)(basep + off);  off = au(off + (size_t)N * 32 * 4);
  float* araw  = (float*)(basep + off);  off = au(off + (size_t)N * 4);
  __bf16* sbf  = (__bf16*)(basep + off); off = au(off + (size_t)N * 64 * 2);

  const int TB = 256;
  const int gpt = (N + TB - 1) / TB;       // per-point grids
  const int gcv = (N + 127) / 128;         // WMMA conv grids (128 pts / block)

  // prep
  zero_kernel<<<2, TB, 0, stream>>>(wsf, 512);
  cvt_bf16_kernel<<<(N * 32 + TB - 1) / TB, TB, 0, stream>>>(feats, fb, N * 32);
  transpose_w_kernel<<<(27 * 32 * 64 + TB - 1) / TB, TB, 0, stream>>>(W1, W1T, 27, 32, 64);
  transpose_w_kernel<<<(27 * 64 * 64 + TB - 1) / TB, TB, 0, stream>>>(W2, W2T, 27, 64, 64);

  // conv1 + BN1 + relu -> x ; SE pool ; att_raw
  conv1_wmma_kernel<<<gcv, TB, 0, stream>>>(fb, nbr1, W1T, y1, sum1, sq1, N);
  finalize_bn_kernel<<<1, 64, 0, stream>>>(sum1, sq1, g1, b1, bn1s, bn1h, 64, invN);
  apply_bn1_kernel<<<gpt, TB, 0, stream>>>(y1, bn1s, bn1h, bidx, Wp1, x, attraw,
                                           pool, asum, asq, N);

  // SE head + att BN
  se_kernel<<<1, 128, 0, stream>>>(pool, Wse1, bse1, Wse2, bse2, sey, 2.0f / (float)N);
  finalize_bn_kernel<<<1, 64, 0, stream>>>(asum, asq, gp1, bp1, atts, atth, 8, invN);
  bn_apply_kernel<<<(int)(((long)N * 8 + TB - 1) / TB), TB, 0, stream>>>(
      attraw, att, atts, atth, 8, (long)N * 8);

  // dilated branch
  dilated_kernel<<<gpt, TB, 0, stream>>>(att, nbr1, nbr2, nbr3, nbr4,
                                         Wd1, Wd2, Wd3, Wd4, draw, dsum, dsq, N);
  for (int d = 0; d < 4; ++d)
    finalize_bn_kernel<<<1, 64, 0, stream>>>(dsum + 8 * d, dsq + 8 * d, gds[d], bds[d],
                                             ds_ + 8 * d, dh_ + 8 * d, 8, invN);
  a_kernel<<<gpt, TB, 0, stream>>>(att, draw, ds_, dh_, Wp2, araw, a2sum, a2sq, N);
  finalize_bn_kernel<<<1, 64, 0, stream>>>(a2sum, a2sq, gp2, bp2, as_, ah_, 1, invN);

  // s = x + se + sp  (as x * (1 + y + a)), bf16 for conv2
  s_kernel<<<gpt, TB, 0, stream>>>(x, araw, as_, ah_, sey, bidx, sbf, N);

  // conv2 + BN2 + relu (in-place on d_out)
  conv2_wmma_kernel<<<gcv, TB, 0, stream>>>(sbf, nbr1, W2T, (float*)d_out, sum2, sq2, N);
  finalize_bn_kernel<<<1, 64, 0, stream>>>(sum2, sq2, g2, b2, bn2s, bn2h, 64, invN);
  bn_apply_kernel<<<(int)(((long)N * 64 + TB - 1) / TB), TB, 0, stream>>>(
      (const float*)d_out, (float*)d_out, bn2s, bn2h, 64, (long)N * 64);
}
